// DeformableCrossAttention_63556926046233
// MI455X (gfx1250) — matrix operand order repaired and checked by reading, not compiled, not run on hardware
//
#include <hip/hip_runtime.h>
#include <hip/hip_bf16.h>
#include <stdint.h>

typedef __attribute__((ext_vector_type(16))) __bf16 v16bf;
typedef __attribute__((ext_vector_type(8)))  float  v8f;
typedef __attribute__((ext_vector_type(4)))  unsigned int v4u;
typedef __attribute__((ext_vector_type(8)))  int v8i;
typedef __attribute__((ext_vector_type(4)))  int v4i;

// --- toolchain probes (read back via compile stderr) ---
#if defined(__has_builtin) && __has_builtin(__builtin_amdgcn_tensor_load_to_lds)
#warning "PROBE: tensor_load_to_lds builtin AVAILABLE (TDM path active)"
#else
#warning "PROBE: tensor_load_to_lds builtin NOT available (fallback copy active)"
#endif
#if defined(__has_builtin) && __has_builtin(__builtin_amdgcn_s_wait_tensorcnt)
#warning "PROBE: s_wait_tensorcnt builtin AVAILABLE"
#endif

#define NB       4
#define LQ       2048
#define LIN      16464
#define DMODEL   256
#define DHEAD    64
#define NHEADS   6
#define NLEVELS  3
#define NPOINTS  9
#define COLS_VAL (NHEADS * DHEAD)                     // 384
#define COLS_OFF (NHEADS * NLEVELS * NPOINTS * 2)     // 324
#define COLS_ATT (NHEADS * NLEVELS * NPOINTS)         // 162
#define COLS_QPR (COLS_OFF + COLS_ATT)                // 486
#define COLS_QP  512                                  // padded so N%64==0

// ---------------------------------------------------------------------------
// Prep kernels: fp32 -> bf16 conversions / packing
// ---------------------------------------------------------------------------
__global__ void cvt_f32_bf16_kernel(const float* __restrict__ in,
                                    __bf16* __restrict__ out, int n) {
  int i = blockIdx.x * blockDim.x + threadIdx.x;
  if (i < n) out[i] = (__bf16)in[i];
}

__global__ void add_cvt_bf16_kernel(const float* __restrict__ a,
                                    const float* __restrict__ b,
                                    __bf16* __restrict__ out, int n) {
  int i = blockIdx.x * blockDim.x + threadIdx.x;
  if (i < n) out[i] = (__bf16)(a[i] + b[i]);
}

// Wt[nc*K + k] = W[k*ncols + nc]  (N-major bf16 so B-fragment K-runs are contiguous)
__global__ void transpose_w_kernel(const float* __restrict__ W,
                                   __bf16* __restrict__ Wt, int K, int ncols) {
  int idx = blockIdx.x * blockDim.x + threadIdx.x;
  if (idx < K * ncols) {
    int k = idx / ncols, nc = idx % ncols;
    Wt[(size_t)nc * K + k] = (__bf16)W[(size_t)k * ncols + nc];
  }
}

// Fused, padded [W_off | W_attn] transpose + bias pack (cols 486..511 zeroed)
__global__ void pack_qp_kernel(const float* __restrict__ Woff, const float* __restrict__ boff,
                               const float* __restrict__ Watt, const float* __restrict__ batt,
                               __bf16* __restrict__ Wt, float* __restrict__ bqp) {
  int idx = blockIdx.x * blockDim.x + threadIdx.x;
  if (idx < COLS_QP * DMODEL) {
    int nc = idx / DMODEL, k = idx % DMODEL;
    float w = 0.f;
    if (nc < COLS_OFF)      w = Woff[(size_t)k * COLS_OFF + nc];
    else if (nc < COLS_QPR) w = Watt[(size_t)k * COLS_ATT + (nc - COLS_OFF)];
    Wt[(size_t)nc * DMODEL + k] = (__bf16)w;
  }
  if (idx < COLS_QP) {
    float b = 0.f;
    if (idx < COLS_OFF)      b = boff[idx];
    else if (idx < COLS_QPR) b = batt[idx - COLS_OFF];
    bqp[idx] = b;
  }
}

// ---------------------------------------------------------------------------
// WMMA GEMM: C[M x N] = A[M x K](bf16,row-major) * Bt[N x K](bf16) + bias
// Block = 4 waves; block tile = 64 rows x 64 cols. The 64xK B-panel (32-48KB)
// is staged into LDS with ONE Tensor-Data-Mover tensor_load_to_lds issue
// (TENSORcnt + s_wait_tensorcnt), then each wave computes a 16x64 strip.
// Inner loop: batch all 8 ds_load_b128 (one dscnt wait), then 4 back-to-back
// v_wmma_f32_16x16x32_bf16. Fragment K order: contiguous k = 16*half + s in both A and B.
// ---------------------------------------------------------------------------
template <bool OUT_BF16>
__global__ void gemm_wmma_bf16_kernel(const __bf16* __restrict__ A,
                                      const __bf16* __restrict__ Bt,
                                      const float* __restrict__ bias,
                                      void* __restrict__ Cout,
                                      int M, int Ncols, int K,
                                      const unsigned char* __restrict__ rowmask) {
  extern __shared__ __bf16 smemB[];              // 64 x K bf16 panel at LDS offset 0
  const int lane = threadIdx.x & 31;
  const int wave = threadIdx.x >> 5;
  const int hl   = lane >> 4;                    // lane half
  const int mr   = lane & 15;

  const int nbase = blockIdx.y * 64;             // N-panel start
  const size_t panelBytes = (size_t)64 * K * 2;
  const __bf16* Bpanel = Bt + (size_t)nbase * K;

#if defined(__has_builtin) && __has_builtin(__builtin_amdgcn_tensor_load_to_lds)
  if (wave == 0) {
    // Tensor DMA descriptor (ISA 08 / D# groups): 1-D tile of 8-byte elements.
    const uint64_t ga   = (uint64_t)(uintptr_t)Bpanel;
    const uint32_t nel8 = (uint32_t)(panelBytes >> 3);     // #8B elements (<=6144)
    v4u g0 = {};
    g0.x = 1u;                                             // count=1 (valid, user)
    g0.y = 0u;                                             // lds_addr = 0 (dyn-LDS base)
    g0.z = (uint32_t)ga;                                   // global_addr[31:0]
    g0.w = (uint32_t)((ga >> 32) & 0x01FFFFFFu) | 0x80000000u; // addr[56:32] | type=2
    v8i g1 = {};
    g1[0] = 0x30000;                                       // data_size=3 (8B), mask=0
    g1[1] = (int)((nel8 & 0xFFFFu) << 16);                 // tensor_dim0[15:0]
    g1[2] = (int)(((nel8 >> 16) & 0xFFFFu) | (1u << 16));  // tdim0 hi | tensor_dim1=1
    g1[3] = (int)((nel8 & 0xFFFFu) << 16);                 // tile_dim0 = nel8
    g1[4] = 1;                                             // tile_dim1=1, tile_dim2=0
    g1[5] = (int)nel8;                                     // tensor_dim0_stride[31:0]
    g1[6] = (int)((nel8 & 0xFFFFu) << 16);                 // | tensor_dim1_stride lo
    g1[7] = (int)(nel8 >> 16);                             // tensor_dim1_stride hi
    v4i g2 = {}, g3 = {};
#if __clang_major__ >= 23
    v8i g4 = {};
    __builtin_amdgcn_tensor_load_to_lds(g0, g1, g2, g3, g4, 0);
#else
    __builtin_amdgcn_tensor_load_to_lds(g0, g1, g2, g3, 0);
#endif
#if __has_builtin(__builtin_amdgcn_s_wait_tensorcnt)
    __builtin_amdgcn_s_wait_tensorcnt(0);
#else
    asm volatile("s_wait_tensorcnt 0x0" ::: "memory");
#endif
  }
#else
  {
    const uint4* gsrc = (const uint4*)Bpanel;
    uint4* ldst = (uint4*)smemB;
    const int nchunks = (int)(panelBytes >> 4);
    for (int i = threadIdx.x; i < nchunks; i += blockDim.x) ldst[i] = gsrc[i];
  }
#endif
  __syncthreads();

  const int tile_m = blockIdx.x * 4 + wave;      // wave's 16-row tile
  const __bf16* Arow = A + (size_t)(tile_m * 16 + mr) * K;
  const __bf16* Bfrag = smemB + (size_t)mr * K + 16 * hl;  // ns strides by 16*K

  v8f acc[4] = {{}, {}, {}, {}};
  for (int kb = 0; kb < K; kb += 32) {
    union { v16bf v; uint4 q[2]; } a;
    // A 16x32 bf16: u[0..7] hold K=16*half+0..15 (matches B)
    a.q[0] = *(const uint4*)(Arow + kb + 16 * hl);
    a.q[1] = *(const uint4*)(Arow + kb + 16 * hl + 8);

    // Batch ALL B-fragment loads first (single dscnt wait, ds clause)
    union { v16bf v; uint4 q[2]; } b[4];
#pragma unroll
    for (int ns = 0; ns < 4; ++ns) {
      // B 32x16 bf16: u[j] holds K = 16*half + 2j (contiguous 32B run in LDS)
      const __bf16* bp = Bfrag + (size_t)ns * 16 * K + kb;
      b[ns].q[0] = *(const uint4*)(bp);
      b[ns].q[1] = *(const uint4*)(bp + 8);
    }
    // Then 4 back-to-back WMMAs (independent accumulators, no D->A/B hazard)
#pragma unroll
    for (int ns = 0; ns < 4; ++ns) {
      acc[ns] = __builtin_amdgcn_wmma_f32_16x16x32_bf16(false, a.v, false, b[ns].v,
                                                        (short)0, acc[ns], false, false);
    }
  }

#pragma unroll
  for (int ns = 0; ns < 4; ++ns) {
    const int ncol = nbase + ns * 16 + mr;
    const float bv = bias ? bias[ncol] : 0.f;
#pragma unroll
    for (int r = 0; r < 8; ++r) {
      const int row = tile_m * 16 + r + 8 * hl;
      float o = acc[ns][r] + bv;
      if (rowmask && rowmask[row]) o = 0.f;      // padding-mask epilogue (value GEMM)
      if (OUT_BF16) ((__bf16*)Cout)[(size_t)row * Ncols + ncol] = (__bf16)o;
      else          ((float*) Cout)[(size_t)row * Ncols + ncol] = o;
    }
  }
}

// ---------------------------------------------------------------------------
// Deformable sampling: 1 block per (n,q); wave = head; lane owns dims {l, l+32}
// value (bf16, ~50MB) is L2-resident on MI455X (192MB L2) -> gathers hit L2.
// ---------------------------------------------------------------------------
__global__ void sample_kernel(const float* __restrict__ refpts,  // (N,Lq,L,2)
                              const int*   __restrict__ shapes,  // (L,2) h,w
                              const int*   __restrict__ lstart,  // (L)
                              const __bf16* __restrict__ value,  // (N*LIN, 384)
                              const float* __restrict__ qproj,   // (N*Lq, 512)
                              __bf16* __restrict__ headout) {    // (N*Lq, 384)
  const int nq   = blockIdx.x;
  const int n    = nq / LQ;
  const int h    = threadIdx.x >> 5;
  const int lane = threadIdx.x & 31;

  const float* qp     = qproj + (size_t)nq * COLS_QP;
  const float* logits = qp + COLS_OFF + h * (NLEVELS * NPOINTS);

  float mx = -INFINITY;
#pragma unroll
  for (int i = 0; i < NLEVELS * NPOINTS; ++i) mx = fmaxf(mx, logits[i]);
  float s = 0.f;
#pragma unroll
  for (int i = 0; i < NLEVELS * NPOINTS; ++i) s += __expf(logits[i] - mx);
  const float inv_s = 1.f / s;

  float acc0 = 0.f, acc1 = 0.f;
  const int d0 = lane, d1 = lane + 32;

  for (int l = 0; l < NLEVELS; ++l) {
    const int   H_l = shapes[2 * l], W_l = shapes[2 * l + 1];
    const int   st  = lstart[l];
    const float rx  = refpts[((size_t)nq * NLEVELS + l) * 2 + 0];
    const float ry  = refpts[((size_t)nq * NLEVELS + l) * 2 + 1];
    const float* offl = qp + (size_t)(h * NLEVELS + l) * NPOINTS * 2;
#pragma unroll
    for (int p = 0; p < NPOINTS; ++p) {
      const float aw = __expf(logits[l * NPOINTS + p] - mx) * inv_s;
      const float lx = rx + offl[2 * p + 0] / (float)W_l;
      const float ly = ry + offl[2 * p + 1] / (float)H_l;
      const float x = lx * (float)W_l - 0.5f, y = ly * (float)H_l - 0.5f;
      const float xf = floorf(x), yf = floorf(y);
      const float wx1 = x - xf, wy1 = y - yf, wx0 = 1.f - wx1, wy0 = 1.f - wy1;
      const int x0 = (int)xf, y0 = (int)yf;
#pragma unroll
      for (int t = 0; t < 4; ++t) {
        const int xi = x0 + (t & 1), yi = y0 + (t >> 1);
        const float w = (t & 1 ? wx1 : wx0) * (t >> 1 ? wy1 : wy0);
        if (xi >= 0 && xi < W_l && yi >= 0 && yi < H_l) {
          const __bf16* vp = value +
              ((size_t)(n * LIN + st + yi * W_l + xi)) * COLS_VAL + h * DHEAD;
          const float g = aw * w;
          acc0 += g * (float)vp[d0];
          acc1 += g * (float)vp[d1];
        }
      }
    }
  }
  __bf16* op = headout + (size_t)nq * COLS_VAL + h * DHEAD;
  op[d0] = (__bf16)acc0;
  op[d1] = (__bf16)acc1;
}

// ---------------------------------------------------------------------------
// Host launcher
// ---------------------------------------------------------------------------
extern "C" void kernel_launch(void* const* d_in, const int* in_sizes, int n_in,
                              void* d_out, int out_size, void* d_ws, size_t ws_size,
                              hipStream_t stream) {
  (void)in_sizes; (void)n_in; (void)out_size; (void)ws_size;
  const float* tgt    = (const float*)d_in[0];
  const float* src    = (const float*)d_in[1];
  const float* qpos   = (const float*)d_in[2];
  const float* refpts = (const float*)d_in[3];
  const int*   shapes = (const int*)d_in[4];
  const int*   lstart = (const int*)d_in[5];
  const unsigned char* padmask = (const unsigned char*)d_in[6];
  const float* W_off  = (const float*)d_in[7];
  const float* b_off  = (const float*)d_in[8];
  const float* W_att  = (const float*)d_in[9];
  const float* b_att  = (const float*)d_in[10];
  const float* W_val  = (const float*)d_in[11];
  const float* b_val  = (const float*)d_in[12];
  const float* W_out  = (const float*)d_in[13];
  const float* b_out  = (const float*)d_in[14];
  float* out = (float*)d_out;

  // workspace carve (256B aligned)
  char* wp = (char*)d_ws;
  auto carve = [&](size_t bytes) {
    char* p = wp; wp += (bytes + 255) & ~(size_t)255; return p;
  };
  __bf16* src_bf  = (__bf16*)carve((size_t)NB * LIN * DMODEL * 2);   // 33.7 MB
  __bf16* q_bf    = (__bf16*)carve((size_t)NB * LQ  * DMODEL * 2);   //  4.2 MB
  __bf16* val_bf  = (__bf16*)carve((size_t)NB * LIN * COLS_VAL * 2); // 50.6 MB
  float*  qproj   = (float*) carve((size_t)NB * LQ  * COLS_QP * 4);  // 16.8 MB
  __bf16* head_bf = (__bf16*)carve((size_t)NB * LQ  * COLS_VAL * 2); //  6.3 MB
  __bf16* Wval_t  = (__bf16*)carve((size_t)COLS_VAL * DMODEL * 2);
  __bf16* Wqp_t   = (__bf16*)carve((size_t)COLS_QP  * DMODEL * 2);
  float*  bqp     = (float*) carve((size_t)COLS_QP * 4);
  __bf16* Wout_t  = (__bf16*)carve((size_t)DMODEL * COLS_VAL * 2);

  const int TB = 256;
  // 1) conversions & weight packing
  {
    int n = NB * LIN * DMODEL;
    cvt_f32_bf16_kernel<<<(n + TB - 1) / TB, TB, 0, stream>>>(src, src_bf, n);
  }
  {
    int n = NB * LQ * DMODEL;
    add_cvt_bf16_kernel<<<(n + TB - 1) / TB, TB, 0, stream>>>(tgt, qpos, q_bf, n);
  }
  {
    int n = DMODEL * COLS_VAL;
    transpose_w_kernel<<<(n + TB - 1) / TB, TB, 0, stream>>>(W_val, Wval_t, DMODEL, COLS_VAL);
    transpose_w_kernel<<<(n + TB - 1) / TB, TB, 0, stream>>>(W_out, Wout_t, COLS_VAL, DMODEL);
  }
  {
    int n = COLS_QP * DMODEL;
    pack_qp_kernel<<<(n + TB - 1) / TB, TB, 0, stream>>>(W_off, b_off, W_att, b_att, Wqp_t, bqp);
  }

  // 2) value = src @ W_val + b_val (bf16 out, padding-mask epilogue)
  {
    const int M = NB * LIN, N = COLS_VAL, K = DMODEL;
    dim3 grid(M / 64, N / 64);
    gemm_wmma_bf16_kernel<true><<<grid, 128, (size_t)64 * K * 2, stream>>>(
        src_bf, Wval_t, b_val, (void*)val_bf, M, N, K, padmask);
  }
  // 3) qproj = q @ [W_off|W_attn] + [b_off|b_attn]  (f32 out, padded N=512)
  {
    const int M = NB * LQ, N = COLS_QP, K = DMODEL;
    dim3 grid(M / 64, N / 64);
    gemm_wmma_bf16_kernel<false><<<grid, 128, (size_t)64 * K * 2, stream>>>(
        q_bf, Wqp_t, bqp, (void*)qproj, M, N, K, nullptr);
  }
  // 4) softmax + bilinear sampling + head accumulate
  sample_kernel<<<NB * LQ, NHEADS * 32, 0, stream>>>(
      refpts, shapes, lstart, val_bf, qproj, head_bf);
  // 5) out = head_out @ W_out + b_out  (f32 to d_out)
  {
    const int M = NB * LQ, N = DMODEL, K = COLS_VAL;
    dim3 grid(M / 64, N / 64);
    gemm_wmma_bf16_kernel<false><<<grid, 128, (size_t)64 * K * 2, stream>>>(
        head_bf, Wout_t, b_out, (void*)out, M, N, K, nullptr);
  }
}
